// Differentiable_EXPHYDRO_23742579212558
// MI455X (gfx1250) — matrix operand order repaired and checked by reading, not compiled, not run on hardware
//
#include <hip/hip_runtime.h>
#include <hip/hip_bf16.h>

typedef _Float16 half_t;
typedef __attribute__((ext_vector_type(16))) _Float16 v16h;
typedef __attribute__((ext_vector_type(8)))  _Float16 v8h;
typedef __attribute__((ext_vector_type(8)))  float    v8f;

#define B_DIM 128
#define T_DIM 4096
#define NROWS (B_DIM * T_DIM)   // 524288
#define CLIPV 100000.0f
#define H1PITCH 264             // halfs; 528B = 33*16 -> b128-aligned slices
#define H2PITCH 72              // halfs; 144B = 9*16

// Fast approximate reciprocal: single v_rcp_f32 (NOT the IEEE div sequence)
__device__ __forceinline__ float fast_rcp(float x) {
    return __builtin_amdgcn_rcpf(x);
}

// Hardware tanh if the toolchain exposes it (gfx1250 TRANS op), else exp+rcp
__device__ __forceinline__ float fast_tanh(float x) {
#if __has_builtin(__builtin_amdgcn_tanhf)
    return __builtin_amdgcn_tanhf(x);
#elif __has_builtin(__builtin_amdgcn_tanh_f32)
    return __builtin_amdgcn_tanh_f32(x);
#else
    return 1.0f - 2.0f * fast_rcp(1.0f + __expf(2.0f * x));
#endif
}

// heaviside(x) = (tanh(5x)+1)/2  (== sigmoid(10x))
__device__ __forceinline__ float hv(float x) {
#if __has_builtin(__builtin_amdgcn_tanhf) || __has_builtin(__builtin_amdgcn_tanh_f32)
    return fmaf(fast_tanh(5.0f * x), 0.5f, 0.5f);
#else
    return fast_rcp(1.0f + __expf(-10.0f * x));
#endif
}

__device__ __forceinline__ float sigmoidf(float x) {
    return fast_rcp(1.0f + __expf(-x));
}

// ---------------------------------------------------------------------------
// Pack w1/w2/w3 (f32, row-major KxN) into f16 B-fragment layout:
// element e of lane l in block (kk,nt): K = kk*32 + (l>=16?8:0) + (e<8?e:e+8),
// N = nt*16 + (l&15). Each fragment is 32 lanes * 16 halfs contiguous.
// ---------------------------------------------------------------------------
__global__ void pack_weights_kernel(const float* __restrict__ w1,
                                    const float* __restrict__ w2,
                                    const float* __restrict__ w3,
                                    half_t* __restrict__ pW1,
                                    half_t* __restrict__ pW2,
                                    half_t* __restrict__ pW3) {
    int idx = blockIdx.x * 256 + threadIdx.x;
    if (idx < 8192) {  // w1: (15,256) -> 16 n-tiles, K padded 15->32
        int nt = idx >> 9, rem = idx & 511, lane = rem >> 4, e = rem & 15;
        int K = ((lane < 16) ? 0 : 8) + ((e < 8) ? e : e + 8);
        int N = nt * 16 + (lane & 15);
        pW1[idx] = (K < 15) ? (half_t)w1[K * 256 + N] : (half_t)0.0f;
    }
    if (idx < 16384) { // w2: (256,64) -> blocks (kk 0..7, n2 0..3)
        int blk = idx >> 9, rem = idx & 511, lane = rem >> 4, e = rem & 15;
        int kk = blk >> 2, n2 = blk & 3;
        int K = kk * 32 + ((lane < 16) ? 0 : 8) + ((e < 8) ? e : e + 8);
        int N = n2 * 16 + (lane & 15);
        pW2[idx] = (half_t)w2[K * 64 + N];
    }
    if (idx < 1024) {  // w3: (64,6) -> kk 0..1, N padded 6->16
        int kk = idx >> 9, rem = idx & 511, lane = rem >> 4, e = rem & 15;
        int K = kk * 32 + ((lane < 16) ? 0 : 8) + ((e < 8) ? e : e + 8);
        int n = lane & 15;
        pW3[idx] = (n < 6) ? (half_t)w3[K * 6 + n] : (half_t)0.0f;
    }
}

// ---------------------------------------------------------------------------
// Fused 3-layer MLP. One wave per 16-row tile; 4 waves/block.
// Layer1: 16 WMMAs (K=32, 15 real), Layer2: 32 WMMAs (K=256), Layer3: 2 WMMAs.
// Activations transposed D-layout -> A-layout via per-wave LDS staging.
// ---------------------------------------------------------------------------
__global__ __launch_bounds__(128) void mlp_kernel(
    const float* __restrict__ in,
    const float* __restrict__ b1, const float* __restrict__ b2,
    const float* __restrict__ b3,
    const half_t* __restrict__ pW1, const half_t* __restrict__ pW2,
    const half_t* __restrict__ pW3,
    float* __restrict__ params) {
    __shared__ half_t lds_h1[4 * 16 * H1PITCH];
    __shared__ half_t lds_h2[4 * 16 * H2PITCH];

    const int tid  = threadIdx.x;
    const int wave = tid >> 5;
    const int lane = tid & 31;
    const int lrow = lane & 15;      // N-col within tile / M-row for A frags
    const int lhi  = lane >> 4;
    const int klo  = lhi * 8;        // A-fragment K skew for upper half-wave
    const int tile = blockIdx.x * 4 + wave;

    half_t* h1 = lds_h1 + wave * 16 * H1PITCH;
    half_t* h2 = lds_h2 + wave * 16 * H2PITCH;

    const v16h* w1f = (const v16h*)pW1;
    const v16h* w2f = (const v16h*)pW2;
    const v16h* w3f = (const v16h*)pW3;

    // ---- A1 fragment: 16x32 f16, features 5..19 of this row, K>=15 zero ----
    const long row = (long)tile * 16 + lrow;
    const float* arow = in + row * 20 + 5;
    v16h a1;
#pragma unroll
    for (int e = 0; e < 16; ++e) {
        int K = ((e < 8) ? e : e + 8) + klo;   // A-layout K mapping
        a1[e] = (K < 15) ? (half_t)arow[K] : (half_t)0.0f;
    }

    // ---- Layer 1: h1 = tanh(attrs @ w1 + b1), 16 N-tiles of 16 ----
#pragma unroll
    for (int nt = 0; nt < 16; ++nt) {
        v8f acc = {};
        acc = __builtin_amdgcn_wmma_f32_16x16x32_f16(
            false, a1, false, w1f[nt * 32 + lane], (short)0, acc, false, false);
        float bias = b1[nt * 16 + lrow];
#pragma unroll
        for (int v = 0; v < 8; ++v) {
            float hval = fast_tanh(acc[v] + bias);
            h1[(v + lhi * 8) * H1PITCH + nt * 16 + lrow] = (half_t)hval;
        }
    }

    // ---- Layer 2: h2 = tanh(h1 @ w2 + b2); K=256 in 8 steps, N=64 ----
    v8f acc2[4] = {{}, {}, {}, {}};
#pragma unroll
    for (int kk = 0; kk < 8; ++kk) {
        const half_t* base = h1 + lrow * H1PITCH + kk * 32 + klo;
        v8h lo = *(const v8h*)(base);
        v8h hi = *(const v8h*)(base + 16);
        v16h a2 = __builtin_shufflevector(lo, hi, 0, 1, 2, 3, 4, 5, 6, 7,
                                          8, 9, 10, 11, 12, 13, 14, 15);
#pragma unroll
        for (int n2 = 0; n2 < 4; ++n2) {
            acc2[n2] = __builtin_amdgcn_wmma_f32_16x16x32_f16(
                false, a2, false, w2f[(kk * 4 + n2) * 32 + lane], (short)0,
                acc2[n2], false, false);
        }
    }
#pragma unroll
    for (int n2 = 0; n2 < 4; ++n2) {
        float bias = b2[n2 * 16 + lrow];
#pragma unroll
        for (int v = 0; v < 8; ++v) {
            float hval = fast_tanh(acc2[n2][v] + bias);
            h2[(v + lhi * 8) * H2PITCH + n2 * 16 + lrow] = (half_t)hval;
        }
    }

    // ---- Layer 3: params = sigmoid(h2 @ w3 + b3); K=64, N=6 (padded 16) ----
    v8f acc3 = {};
#pragma unroll
    for (int kk = 0; kk < 2; ++kk) {
        const half_t* base = h2 + lrow * H2PITCH + kk * 32 + klo;
        v8h lo = *(const v8h*)(base);
        v8h hi = *(const v8h*)(base + 16);
        v16h a3 = __builtin_shufflevector(lo, hi, 0, 1, 2, 3, 4, 5, 6, 7,
                                          8, 9, 10, 11, 12, 13, 14, 15);
        acc3 = __builtin_amdgcn_wmma_f32_16x16x32_f16(
            false, a3, false, w3f[kk * 32 + lane], (short)0, acc3, false, false);
    }
    if (lrow < 6) {
        float bias = b3[lrow];
#pragma unroll
        for (int v = 0; v < 8; ++v) {
            long r = (long)tile * 16 + v + lhi * 8;
            params[r * 6 + lrow] = sigmoidf(acc3[v] + bias);
        }
    }
}

// ---------------------------------------------------------------------------
// Sequential bucket scan: one thread per basin (B=128), 4 waves on 4 WGPs.
// Writes post-step s1 history; final soilbucket is done in parallel later.
// ---------------------------------------------------------------------------
__global__ __launch_bounds__(32) void scan_kernel(
    const float* __restrict__ in, const float* __restrict__ params,
    float* __restrict__ s1out) {
    const int b = blockIdx.x * 32 + threadIdx.x;
    float s0 = 0.0f, s1 = 0.0f;
    const float* ip = in + (long)b * T_DIM * 20;
    const float* pp = params + (long)b * T_DIM * 6;
    float* sp = s1out + (long)b * T_DIM;
    for (int t = 0; t < T_DIM; ++t) {
        __builtin_prefetch(ip + 20 * 16, 0, 0);   // speculative, OOB-safe
        __builtin_prefetch(pp + 6 * 16, 0, 0);
        float pet = ip[0], tm = ip[1], pr = ip[2];
        float tmin = pp[0], tmax = pp[1], ddf = pp[2];
        float f = pp[3], smax = pp[4], qmax = pp[5];
        ip += 20; pp += 6;

        float tmn   = tmin * -3.0f;
        float psnow = hv(tmn - tm) * pr;
        float prain = pr - psnow;                  // hv(t-tmn) == 1-hv(tmn-t)
        float tmx3  = tmax * 3.0f;
        float melt  = hv(tm - tmx3) * hv(s0) * fminf(s0, ddf * 5.0f * (tm - tmx3));

        float f2    = f * 0.1f;
        float smax2 = smax * 1400.0f + 100.0f;
        float qmax2 = qmax * 40.0f + 10.0f;
        float h1    = hv(s1);
        float above = hv(s1 - smax2);
        float below = 1.0f - above;
        float et    = h1 * above * pet + h1 * below * pet * (s1 * fast_rcp(smax2));
        float qsub  = h1 * above * qmax2 +
                      h1 * below * qmax2 * __expf(-f2 * (smax2 - s1));
        float qsurf = h1 * above * (s1 - smax2);

        float ds0 = psnow - melt;
        float ds1 = prain + melt - et - qsub - qsurf;
        s0 += fminf(fmaxf(ds0, -CLIPV), CLIPV);
        s1 += fminf(fmaxf(ds1, -CLIPV), CLIPV);
        sp[t] = s1;
    }
}

// ---------------------------------------------------------------------------
// Parallel epilogue: q = qsub + qsurf from s1 history (B*T threads)
// ---------------------------------------------------------------------------
__global__ __launch_bounds__(256) void final_q_kernel(
    const float* __restrict__ in, const float* __restrict__ params,
    const float* __restrict__ s1arr, float* __restrict__ out, int n) {
    int i = blockIdx.x * 256 + threadIdx.x;
    if (i >= n) return;
    float s1   = s1arr[i];
    float f    = params[(long)i * 6 + 3];
    float smax = params[(long)i * 6 + 4];
    float qmax = params[(long)i * 6 + 5];
    float f2 = f * 0.1f;
    float smax2 = smax * 1400.0f + 100.0f;
    float qmax2 = qmax * 40.0f + 10.0f;
    float h1 = hv(s1), above = hv(s1 - smax2), below = 1.0f - above;
    float qsub  = h1 * above * qmax2 +
                  h1 * below * qmax2 * __expf(-f2 * (smax2 - s1));
    float qsurf = h1 * above * (s1 - smax2);
    out[i] = qsub + qsurf;
}

extern "C" void kernel_launch(void* const* d_in, const int* in_sizes, int n_in,
                              void* d_out, int out_size, void* d_ws,
                              size_t ws_size, hipStream_t stream) {
    const float* in = (const float*)d_in[0];
    const float* w1 = (const float*)d_in[1];
    const float* b1 = (const float*)d_in[2];
    const float* w2 = (const float*)d_in[3];
    const float* b2 = (const float*)d_in[4];
    const float* w3 = (const float*)d_in[5];
    const float* b3 = (const float*)d_in[6];
    float* out = (float*)d_out;

    char* ws = (char*)d_ws;
    half_t* pW1 = (half_t*)(ws);                 // 16 KB
    half_t* pW2 = (half_t*)(ws + 16 * 1024);     // 32 KB
    half_t* pW3 = (half_t*)(ws + 48 * 1024);     //  2 KB
    float* params = (float*)(ws + 64 * 1024);    // NROWS*6*4 = 12.6 MB
    float* s1arr  = (float*)(ws + 64 * 1024 + (size_t)NROWS * 6 * 4); // 2.1 MB

    pack_weights_kernel<<<64, 256, 0, stream>>>(w1, w2, w3, pW1, pW2, pW3);
    mlp_kernel<<<NROWS / 64, 128, 0, stream>>>(in, b1, b2, b3, pW1, pW2, pW3,
                                               params);
    scan_kernel<<<4, 32, 0, stream>>>(in, params, s1arr);
    final_q_kernel<<<(NROWS + 255) / 256, 256, 0, stream>>>(in, params, s1arr,
                                                            out, NROWS);
}
